// NTN_1460288880957
// MI455X (gfx1250) — matrix operand order repaired and checked by reading, not compiled, not run on hardware
//
#include <hip/hip_runtime.h>

#define DD 128   // embedding dim
#define QQ 256
#define MM 256
#define KK 16
#define BB 32
#define PD 136   // padded LDS row stride (bf16 elements), 272B = 16B aligned
#define QT 128   // q rows per workgroup
#define MT 128   // m cols per tile

typedef __bf16 v16bf __attribute__((ext_vector_type(16)));
typedef float  v8f   __attribute__((ext_vector_type(8)));
typedef unsigned int u32x4 __attribute__((ext_vector_type(4)));

union FragU { v16bf v; u32x4 q[2]; };

__device__ __forceinline__ unsigned short f2bf(float f) {
  unsigned int u = __float_as_uint(f);
  u += 0x7FFFu + ((u >> 16) & 1u);          // round-to-nearest-even
  return (unsigned short)(u >> 16);
}

// A fragment: 16(M)x32(K) bf16. lane&15 = row; chunks at K = kc*32 + 8*(lane>>4), +16.
__device__ __forceinline__ v16bf load_frag_a(const unsigned short* base, int rowBase, int kc, int lane) {
  const int row = rowBase + (lane & 15);
  const int kb  = kc * 32 + ((lane >> 4) << 3);
  const unsigned short* p = base + row * PD + kb;
  FragU f;
  f.q[0] = *(const u32x4*)(p);
  f.q[1] = *(const u32x4*)(p + 16);
  return f.v;
}

// B fragment: 32(K)x16(N) bf16 with source stored N-major (row n holds K contiguous).
// lane&15 = n; chunks at K = kc*32 + 8*(lane>>4), +16.
__device__ __forceinline__ v16bf load_frag_b(const unsigned short* base, int colBase, int kc, int lane) {
  const int n  = colBase + (lane & 15);
  const int kb = kc * 32 + ((lane >> 4) << 3);
  const unsigned short* p = base + n * PD + kb;
  FragU f;
  f.q[0] = *(const u32x4*)(p);
  f.q[1] = *(const u32x4*)(p + 16);
  return f.v;
}

extern "C" __global__ __launch_bounds__(256)
void ntn_bilinear_kernel(const float* __restrict__ eq,
                         const float* __restrict__ eda,
                         const float* __restrict__ w,
                         const float* __restrict__ V,
                         const float* __restrict__ bvec,
                         float* __restrict__ out)
{
  extern __shared__ unsigned short smem[];
  unsigned short* sWE = smem;                  // stage1: W^T ; stage2: Eda tile
  unsigned short* sA  = smem + 128 * PD;       // bf16 embed_q tile
  unsigned short* sT  = smem + 2 * 128 * PD;   // bf16 T = Eq @ W
  float* svq = (float*)(smem + 3 * 128 * PD);  // 128 floats
  float* svd = svq + 128;                      // 128 floats

  const int qt = blockIdx.x;   // 0..1  (q half)
  const int k  = blockIdx.y;   // 0..15
  const int b  = blockIdx.z;   // 0..31
  const int t    = threadIdx.x;
  const int wave = t >> 5;
  const int lane = t & 31;

  const float bias = bvec[k];
  const float* wk  = w  + (size_t)k * DD * DD;
  const float* eqb = eq + ((size_t)b * QQ + (size_t)qt * QT) * DD;
  const float* vk  = V  + (size_t)k * 2 * DD;

  // ---- phase 0: stage W^T (for contiguous B-fragment reads), bf16 Eq tile, vq ----
  #pragma unroll 4
  for (int idx = t; idx < DD * DD; idx += 256) {
    const int d = idx >> 7, e = idx & 127;
    sWE[e * PD + d] = f2bf(wk[idx]);           // transpose: sWE[e][d] = W[d][e]
  }
  #pragma unroll 4
  for (int idx = t; idx < QT * DD; idx += 256) {
    sA[(idx >> 7) * PD + (idx & 127)] = f2bf(eqb[idx]);
  }
  if (t < QT) {
    const float* r = eqb + t * DD;
    float s = 0.f;
    #pragma unroll 8
    for (int d = 0; d < DD; ++d) s += vk[d] * r[d];
    svq[t] = s;                                // vq[q] = V[k,:D] . eq[q]
  }
  __syncthreads();

  // ---- stage 1: T = Eq_tile @ W[k]  (wave w: rows [16w,16w+16), cols 0..127) ----
  {
    v8f acc[8];
    const v8f z = {0.f,0.f,0.f,0.f,0.f,0.f,0.f,0.f};
    #pragma unroll
    for (int ct = 0; ct < 8; ++ct) acc[ct] = z;
    #pragma unroll
    for (int kc = 0; kc < 4; ++kc) {
      const v16bf a = load_frag_a(sA, wave * 16, kc, lane);
      #pragma unroll
      for (int ct = 0; ct < 8; ++ct) {
        const v16bf bb = load_frag_b(sWE, ct * 16, kc, lane);
        acc[ct] = __builtin_amdgcn_wmma_f32_16x16x32_bf16(
            false, a, false, bb, (short)0, acc[ct], false, false);
      }
    }
    #pragma unroll
    for (int ct = 0; ct < 8; ++ct) {
      const int col = ct * 16 + (lane & 15);
      #pragma unroll
      for (int i = 0; i < 8; ++i) {
        const int row = wave * 16 + i + ((lane >> 4) << 3);
        sT[row * PD + col] = f2bf(acc[ct][i]);
      }
    }
  }
  __syncthreads();

  const size_t obase = (((size_t)b * KK + k) * QQ + (size_t)qt * QT) * MM;

  // ---- stage 2: loop m tiles; out = sigmoid(T @ Eda^T + vq + vd + bias) ----
  for (int mt = 0; mt < MM / MT; ++mt) {
    if (mt) __syncthreads();                   // prior tile's reads of sWE done
    const float* edb = eda + ((size_t)b * MM + (size_t)mt * MT) * DD;
    #pragma unroll 4
    for (int idx = t; idx < MT * DD; idx += 256) {
      sWE[(idx >> 7) * PD + (idx & 127)] = f2bf(edb[idx]);
    }
    if (t < MT) {
      const float* r = edb + t * DD;
      float s = 0.f;
      #pragma unroll 8
      for (int d = 0; d < DD; ++d) s += vk[DD + d] * r[d];
      svd[t] = s;                              // vd[m] = V[k,D:] . eda[m]
    }
    __syncthreads();

    v8f acc[8];
    const v8f z = {0.f,0.f,0.f,0.f,0.f,0.f,0.f,0.f};
    #pragma unroll
    for (int ct = 0; ct < 8; ++ct) acc[ct] = z;
    #pragma unroll
    for (int kc = 0; kc < 4; ++kc) {
      const v16bf a = load_frag_a(sT, wave * 16, kc, lane);
      #pragma unroll
      for (int ct = 0; ct < 8; ++ct) {
        const v16bf bb = load_frag_b(sWE, ct * 16, kc, lane);
        acc[ct] = __builtin_amdgcn_wmma_f32_16x16x32_bf16(
            false, a, false, bb, (short)0, acc[ct], false, false);
      }
    }

    float rvq[8];
    #pragma unroll
    for (int i = 0; i < 8; ++i)
      rvq[i] = svq[wave * 16 + i + ((lane >> 4) << 3)];

    #pragma unroll
    for (int ct = 0; ct < 8; ++ct) {
      const int col = ct * 16 + (lane & 15);
      const float vdb = svd[col] + bias;
      #pragma unroll
      for (int i = 0; i < 8; ++i) {
        const int row = wave * 16 + i + ((lane >> 4) << 3);
        const float x = acc[ct][i] + rvq[i] + vdb;
        // Output is write-once streaming data (134 MB): non-temporal store so the
        // L2 keeps the ~9 MB of heavily re-read inputs resident instead.
        __builtin_nontemporal_store(
            1.0f / (1.0f + __expf(-x)),
            &out[obase + (size_t)row * MM + (size_t)(mt * MT + col)]);
      }
    }
  }
}

extern "C" void kernel_launch(void* const* d_in, const int* in_sizes, int n_in,
                              void* d_out, int out_size, void* d_ws, size_t ws_size,
                              hipStream_t stream) {
  const float* eq  = (const float*)d_in[0];
  const float* eda = (const float*)d_in[1];
  const float* w   = (const float*)d_in[2];
  const float* V   = (const float*)d_in[3];
  const float* bv  = (const float*)d_in[4];
  float* out = (float*)d_out;

  const size_t smem = (size_t)(3 * 128 * PD) * sizeof(unsigned short)
                    + 256 * sizeof(float);   // ~103.5 KB (< 320 KB/WGP)
  (void)hipFuncSetAttribute(reinterpret_cast<const void*>(ntn_bilinear_kernel),
                            hipFuncAttributeMaxDynamicSharedMemorySize, (int)smem);

  dim3 grid(QQ / QT, KK, BB);   // (2, 16, 32) = 1024 workgroups
  ntn_bilinear_kernel<<<grid, 256, smem, stream>>>(eq, eda, w, V, bv, out);
}
